// PointCloudEncoder_43800076485394
// MI455X (gfx1250) — matrix operand order repaired and checked by reading, not compiled, not run on hardware
//
#include <hip/hip_runtime.h>
#include <hip/hip_bf16.h>
#include <stdint.h>

// ---------------------------------------------------------------------------
// PointCloudEncoder for MI455X (gfx1250, wave32, WMMA 16x16x32 bf16)
//
//   k_embed   : f0 = relu(bn(feat @ w_embed^T))            -> bf16 (B,N,128)
//   k_fps     : farthest point sampling (LDS-resident)     -> sel (B,512)
//   k_knn     : top-32 NN per query (reg distances + LDS reduce)
//   k_cvtw    : fp32->bf16 weight conversion (+pad 259->288)
//   k_expand  : build expanded rows (R x 288 bf16) + rel_raw + stat partials
//   k_stats   : deterministic reduction -> global std (ddof=1)
//   k_weights : w = exp(-|rel_n - lc_xyz|/2)
//   k_gemm<..>: WMMA bf16 GEMM, 16x64 tile/wave (4x A reuse), fused epilogues,
//               weight panel staged via GLOBAL_LOAD_ASYNC_TO_LDS (ASYNCcnt)
//   k_final   : max_k + mean_k -> out (B,256,512)
// ---------------------------------------------------------------------------

#define BATCH   8
#define NPTS    8192
#define DIN     16
#define GRP     512
#define KNN     32
#define OUTD    256
#define MIDD    128
#define PREVD   128
#define CEXP    259
#define CPAD    288              // 259 padded to multiple of 32 for K-loop
#define RROWS   (BATCH*GRP*KNN)  // 131072
#define INVS    0.99999500003749969f  // 1/sqrt(1+1e-5)

typedef __attribute__((ext_vector_type(16))) __bf16 v16bf;
typedef __attribute__((ext_vector_type(8)))  float  v8f;

// ---- CDNA5 async global->LDS path (ASYNCcnt) ------------------------------
// Probe-learned signature: arg0 = AS(1) pointer to 16B int vector (b128),
// arg1 = LDS-side pointer, arg2/arg3 = imm offset / cpol.
#if defined(__has_builtin)
# if __has_builtin(__builtin_amdgcn_global_load_async_to_lds_b128)
#  define HAVE_ASYNC_LOAD 1
# endif
# if __has_builtin(__builtin_amdgcn_s_wait_asynccnt)
#  define HAVE_WAIT_ASYNC 1
# endif
#endif
#if defined(HAVE_ASYNC_LOAD) && defined(HAVE_WAIT_ASYNC)
# define USE_ASYNC_LDS 1
typedef int async_b128 __attribute__((vector_size(16)));
typedef __attribute__((address_space(1))) async_b128* g_b128p;
typedef __attribute__((address_space(3))) async_b128* l_b128p;
#else
# pragma message("gfx1250 async-to-LDS builtins unavailable; using VALU staging")
#endif

// ----------------------------- embed ---------------------------------------
__global__ __launch_bounds__(256) void k_embed(
    const float* __restrict__ feat, const float* __restrict__ w,
    const float* __restrict__ bb,   const float* __restrict__ g,
    const float* __restrict__ be,   __bf16* __restrict__ f0)
{
  __shared__ float sw[PREVD*DIN];
  __shared__ float sb[PREVD], sg[PREVD], sbe[PREVD];
  const int t = threadIdx.x;
  for (int i = t; i < PREVD*DIN; i += 256) sw[i] = w[i];
  for (int i = t; i < PREVD;     i += 256) { sb[i]=bb[i]; sg[i]=g[i]; sbe[i]=be[i]; }
  __syncthreads();
  const long long row = (long long)blockIdx.x * 2 + (t >> 7);   // over B*N
  const int d = t & 127;
  const float* fr = feat + row * DIN;
  float acc = sb[d];
#pragma unroll
  for (int c = 0; c < DIN; ++c) acc += fr[c] * sw[d*DIN + c];
  acc = acc * INVS * sg[d] + sbe[d];
  f0[row*PREVD + d] = (__bf16)fmaxf(acc, 0.f);
}

// ------------------------------ FPS ----------------------------------------
__global__ __launch_bounds__(1024) void k_fps(
    const float* __restrict__ xyz, int* __restrict__ sel)
{
  extern __shared__ float smem[];
  float* sx   = smem;                 // N*3
  float* mind = sx + NPTS*3;          // N
  float* rv   = mind + NPTS;          // 1024
  int*   ri   = (int*)(rv + 1024);    // 1024
  const int b = blockIdx.x, t = threadIdx.x;
  const float* xb = xyz + (long long)b * NPTS * 3;
  for (int i = t; i < NPTS*3; i += 1024) sx[i] = xb[i];
  __syncthreads();
  float bv = -1e30f; int bi = 0;
  for (int i = t; i < NPTS; i += 1024) {
    float x = sx[3*i], y = sx[3*i+1], z = sx[3*i+2];
    float s = x*x + y*y + z*z;
    if (s > bv) { bv = s; bi = i; }
    mind[i] = 1e10f;
  }
  rv[t] = bv; ri[t] = bi; __syncthreads();
  for (int s = 512; s > 0; s >>= 1) {
    if (t < s) {
      if (rv[t+s] > rv[t] || (rv[t+s] == rv[t] && ri[t+s] < ri[t])) { rv[t]=rv[t+s]; ri[t]=ri[t+s]; }
    }
    __syncthreads();
  }
  int cur = ri[0];
  __syncthreads();
  for (int it = 0; it < GRP; ++it) {
    if (t == 0) sel[b*GRP + it] = cur;
    const float cx = sx[3*cur], cy = sx[3*cur+1], cz = sx[3*cur+2];
    float mv = -1e30f; int mi = 0;
    for (int i = t; i < NPTS; i += 1024) {
      float dx = sx[3*i]-cx, dy = sx[3*i+1]-cy, dz = sx[3*i+2]-cz;
      float m = fminf(mind[i], dx*dx + dy*dy + dz*dz);
      mind[i] = m;
      if (m > mv) { mv = m; mi = i; }
    }
    rv[t] = mv; ri[t] = mi; __syncthreads();
    for (int s = 512; s > 0; s >>= 1) {
      if (t < s) {
        if (rv[t+s] > rv[t] || (rv[t+s] == rv[t] && ri[t+s] < ri[t])) { rv[t]=rv[t+s]; ri[t]=ri[t+s]; }
      }
      __syncthreads();
    }
    cur = ri[0];
    __syncthreads();
  }
}

// ------------------------------ KNN ----------------------------------------
__global__ __launch_bounds__(256) void k_knn(
    const float* __restrict__ xyz, const int* __restrict__ sel,
    float* __restrict__ lcx, int* __restrict__ nn_idx)
{
  __shared__ float rv[256]; __shared__ int ri[256]; __shared__ float sq[3];
  const int bg = blockIdx.x;          // b*512+g
  const int b  = bg >> 9;
  const int t  = threadIdx.x;
  const float* xb = xyz + (long long)b * NPTS * 3;
  if (t == 0) {
    int si = sel[bg];
    sq[0] = xb[3*si]; sq[1] = xb[3*si+1]; sq[2] = xb[3*si+2];
    lcx[bg*3] = sq[0]; lcx[bg*3+1] = sq[1]; lcx[bg*3+2] = sq[2];
  }
  __syncthreads();
  const float qx = sq[0], qy = sq[1], qz = sq[2];
  const float qq = qx*qx + qy*qy + qz*qz;
  float d[32];
#pragma unroll
  for (int j = 0; j < 32; ++j) {
    int i = t + j*256;
    float rx = xb[3*i], ry = xb[3*i+1], rz = xb[3*i+2];
    float rr = rx*rx + ry*ry + rz*rz;
    d[j] = fmaxf(qq + rr - 2.f*(qx*rx + qy*ry + qz*rz), 0.f);
  }
  for (int k = 0; k < KNN; ++k) {
    float bv = 1e30f; int bj = 0;
#pragma unroll
    for (int j = 0; j < 32; ++j) { if (d[j] < bv) { bv = d[j]; bj = j; } }
    rv[t] = bv; ri[t] = t + bj*256; __syncthreads();
    for (int s = 128; s > 0; s >>= 1) {
      if (t < s) {
        if (rv[t+s] < rv[t] || (rv[t+s] == rv[t] && ri[t+s] < ri[t])) { rv[t]=rv[t+s]; ri[t]=ri[t+s]; }
      }
      __syncthreads();
    }
    int win = ri[0];
    if (t == 0) nn_idx[(long long)bg*KNN + k] = win;
    if ((win & 255) == t) d[win >> 8] = 1e30f;  // owner removes winner
    __syncthreads();
  }
}

// ------------------------- weight conversion -------------------------------
__global__ __launch_bounds__(256) void k_cvtw(
    const float* __restrict__ w1, const float* __restrict__ dw,
    const float* __restrict__ uw, __bf16* __restrict__ W1,
    __bf16* __restrict__ Wd, __bf16* __restrict__ Wu)
{
  const int t = blockIdx.x*256 + threadIdx.x;
  if (t < OUTD*CPAD) {
    int m = t / CPAD, c = t % CPAD;
    W1[t] = (__bf16)((c < CEXP) ? w1[m*CEXP + c] : 0.f);
  }
  if (t < 2*MIDD*OUTD) Wd[t] = (__bf16)dw[t];
  if (t < 2*OUTD*MIDD) Wu[t] = (__bf16)uw[t];
}

// --------------------------- expanded rows ---------------------------------
__global__ __launch_bounds__(256) void k_expand(
    const float* __restrict__ xyz, const __bf16* __restrict__ f0,
    const int* __restrict__ sel,   const float* __restrict__ lcx,
    const int* __restrict__ nn_idx, __bf16* __restrict__ E,
    float* __restrict__ rel_raw, float* __restrict__ part)
{
  __shared__ float bs[8], bq[8];
  const int lane = threadIdx.x & 31, wv = threadIdx.x >> 5;
  const long long r = (long long)blockIdx.x * 8 + wv;
  const int bg = (int)(r >> 5);
  const int b  = bg >> 9;
  const int nn = nn_idx[r];
  const int ct = sel[bg];
  const __bf16* fnn = f0 + ((long long)b*NPTS + nn) * PREVD;
  const __bf16* fc  = f0 + ((long long)b*NPTS + ct) * PREVD;
  __bf16* row = E + r * CPAD;
#pragma unroll
  for (int c = lane; c < PREVD; c += 32) { row[c] = fnn[c]; row[131 + c] = fc[c]; }
  if (lane < CPAD - CEXP) row[CEXP + lane] = (__bf16)0.f;
  const float* pn = xyz + ((long long)b*NPTS + nn) * 3;
  const float kx = pn[0], ky = pn[1], kz = pn[2];
  if (lane == 0) {
    row[128] = (__bf16)kx; row[129] = (__bf16)ky; row[130] = (__bf16)kz;
    float rx = kx - lcx[bg*3], ry = ky - lcx[bg*3+1], rz = kz - lcx[bg*3+2];
    rel_raw[r*3] = rx; rel_raw[r*3+1] = ry; rel_raw[r*3+2] = rz;
    bs[wv] = rx + ry + rz;
    bq[wv] = rx*rx + ry*ry + rz*rz;
  }
  __syncthreads();
  if (threadIdx.x == 0) {
    float s = 0.f, q = 0.f;
#pragma unroll
    for (int i = 0; i < 8; ++i) { s += bs[i]; q += bq[i]; }
    part[2*blockIdx.x] = s; part[2*blockIdx.x + 1] = q;
  }
}

__global__ __launch_bounds__(256) void k_stats(
    const float* __restrict__ part, float* __restrict__ stats)
{
  __shared__ float ss[256], sq[256];
  const int t = threadIdx.x;
  float a = 0.f, q = 0.f;
  for (int i = t; i < RROWS/8; i += 256) { a += part[2*i]; q += part[2*i+1]; }
  ss[t] = a; sq[t] = q; __syncthreads();
  for (int s = 128; s > 0; s >>= 1) {
    if (t < s) { ss[t] += ss[t+s]; sq[t] += sq[t+s]; }
    __syncthreads();
  }
  if (t == 0) {
    const float M = (float)(RROWS * 3);
    float mean = ss[0] / M;
    float var  = (sq[0] - M*mean*mean) / (M - 1.f);
    stats[0] = sqrtf(fmaxf(var, 0.f)) + 1e-5f;
  }
}

__global__ __launch_bounds__(256) void k_weights(
    const float* __restrict__ rel_raw, const float* __restrict__ lcx,
    const float* __restrict__ stats, float* __restrict__ wgt)
{
  const long long r = (long long)blockIdx.x*256 + threadIdx.x;
  const int bg = (int)(r >> 5);
  const float inv_std = 1.f / stats[0];
  float rx = rel_raw[r*3]   * inv_std - lcx[bg*3];
  float ry = rel_raw[r*3+1] * inv_std - lcx[bg*3+1];
  float rz = rel_raw[r*3+2] * inv_std - lcx[bg*3+2];
  wgt[r] = expf(-0.5f * sqrtf(rx*rx + ry*ry + rz*rz));
}

// --------------------------- WMMA GEMM core --------------------------------
// Y[r, m] = epilogue( sum_c X[r,c] * W[m,c] )
// One wave -> 16 rows x 64 cols (4 col tiles share one A frag -> 4x A reuse).
// Block = 8 waves -> 128 rows x 64 cols; weight panel 64 x CINP in LDS.
// A frag (ISA 7.12.2, 16-bit A 16x32): lane l -> row (l&15),
//   K = {0..7,16..23} for l<16, {8..15,24..31} for l>=16.
// B frag (32x16): lane l -> col (l&15), K same as A frag.
// D: element v of acc at lane l -> (M = v + 8*(l>=16), N = l&15).
// MODE 0: lin1   (A bf16 E)  y = relu(y*s*g+be) * wgt[r]
// MODE 1: down   (A fp32 x)  y = relu((y+db)*s*g+be)
// MODE 2: up     (A fp32 h)  y = relu((y+ub)*s*g+be + x_old)
template<int CIN, int CINP, int COUT, int MODE>
__global__ __launch_bounds__(256) void k_gemm(
    const __bf16* __restrict__ Abf, const float* __restrict__ Af,
    const __bf16* __restrict__ Wt,
    const float* __restrict__ g,  const float* __restrict__ be,
    const float* __restrict__ bw, // bias (MODE 1/2) or wgt[r] (MODE 0)
    float* __restrict__ Y, const float* __restrict__ Xold)
{
  constexpr int LSTR = CINP + 8;                      // LDS skew (16B)
  __shared__ __attribute__((aligned(16))) __bf16 wp[64 * LSTR];
  const int t = threadIdx.x;
  const int m0 = blockIdx.y * 64;
  // ---- stage 64 x CINP weight panel (16B granules) ----
#if defined(USE_ASYNC_LDS)
  for (int idx = t; idx < 64*(CINP/8); idx += 256) {
    int rrow = idx / (CINP/8), cc = (idx % (CINP/8)) * 8;
    __builtin_amdgcn_global_load_async_to_lds_b128(
        (g_b128p)(const void*)(Wt + (m0 + rrow)*CINP + cc),
        (l_b128p)(void*)&wp[rrow*LSTR + cc], 0, 0);
  }
  __builtin_amdgcn_s_wait_asynccnt(0);
  __syncthreads();
#else
  for (int idx = t; idx < 64*(CINP/8); idx += 256) {
    int rrow = idx / (CINP/8), cc = (idx % (CINP/8)) * 8;
    *(uint4*)&wp[rrow*LSTR + cc] = *(const uint4*)&Wt[(m0 + rrow)*CINP + cc];
  }
  __syncthreads();
#endif

  const int wave = t >> 5, lane = t & 31;
  const int hi = lane >> 4, l15 = lane & 15;
  const long long r0   = ((long long)blockIdx.x * 8 + wave) * 16;
  const long long arow = r0 + l15;
  const int aoff = hi * 8;    // A K sub-offset
  const int boff = hi * 8;    // B K sub-offset
  v8f acc[4] = {};
  for (int k0 = 0; k0 < CINP; k0 += 32) {
    v16bf a;
    if constexpr (MODE == 0) {
      const uint4* pa = (const uint4*)(Abf + arow*CINP + k0 + aoff);
      __builtin_prefetch((const void*)(pa + 4), 0, 3);   // next K step
      union { v16bf v; uint4 q[2]; } au;
      au.q[0] = pa[0]; au.q[1] = pa[2];                  // +16 bf16
      a = au.v;
    } else {
      const float4* pa = (const float4*)(Af + arow*CIN + k0 + aoff);
      __builtin_prefetch((const void*)(pa + 8), 0, 3);
      float4 f0v = pa[0], f1v = pa[1], f2v = pa[4], f3v = pa[5];
      a[0]=(__bf16)f0v.x; a[1]=(__bf16)f0v.y; a[2]=(__bf16)f0v.z; a[3]=(__bf16)f0v.w;
      a[4]=(__bf16)f1v.x; a[5]=(__bf16)f1v.y; a[6]=(__bf16)f1v.z; a[7]=(__bf16)f1v.w;
      a[8]=(__bf16)f2v.x; a[9]=(__bf16)f2v.y; a[10]=(__bf16)f2v.z; a[11]=(__bf16)f2v.w;
      a[12]=(__bf16)f3v.x; a[13]=(__bf16)f3v.y; a[14]=(__bf16)f3v.z; a[15]=(__bf16)f3v.w;
    }
#pragma unroll
    for (int tt = 0; tt < 4; ++tt) {
      const uint4* pb = (const uint4*)&wp[(tt*16 + l15)*LSTR + k0 + boff];
      union { v16bf v; uint4 q[2]; } bu;
      bu.q[0] = pb[0]; bu.q[1] = pb[2];
      acc[tt] = __builtin_amdgcn_wmma_f32_16x16x32_bf16(
          false, a, false, bu.v, (short)0, acc[tt], false, false);
    }
  }

#pragma unroll
  for (int tt = 0; tt < 4; ++tt) {
    const int m = m0 + tt*16 + l15;
#pragma unroll
    for (int v = 0; v < 8; ++v) {
      const long long r = r0 + v + 8*hi;
      float y = acc[tt][v];
      if constexpr (MODE == 0) {
        y = y * INVS * g[m] + be[m];
        Y[r*COUT + m] = fmaxf(y, 0.f) * bw[r];
      } else if constexpr (MODE == 1) {
        y = (y + bw[m]) * INVS * g[m] + be[m];
        Y[r*COUT + m] = fmaxf(y, 0.f);
      } else {
        y = (y + bw[m]) * INVS * g[m] + be[m] + Xold[r*COUT + m];
        Y[r*COUT + m] = fmaxf(y, 0.f);
      }
    }
  }
}

// ----------------------------- final reduce --------------------------------
__global__ __launch_bounds__(256) void k_final(
    const float* __restrict__ x, float* __restrict__ out)
{
  const int bg = blockIdx.x, m = threadIdx.x;
  const long long base = (long long)bg * KNN * OUTD;
  float mx = -1e30f, sm = 0.f;
#pragma unroll
  for (int k = 0; k < KNN; ++k) {
    float v = x[base + (long long)k*OUTD + m];
    mx = fmaxf(mx, v); sm += v;
  }
  const int b = bg >> 9, gg = bg & (GRP-1);
  out[((long long)(b*OUTD + m))*GRP + gg] = mx + sm * (1.f/KNN);
}

// ----------------------------- launcher ------------------------------------
extern "C" void kernel_launch(void* const* d_in, const int* in_sizes, int n_in,
                              void* d_out, int out_size, void* d_ws, size_t ws_size,
                              hipStream_t stream) {
  const float* xyz     = (const float*)d_in[0];
  const float* feat    = (const float*)d_in[1];
  const float* w_embed = (const float*)d_in[2];
  const float* b_embed = (const float*)d_in[3];
  const float* g_embed = (const float*)d_in[4];
  const float* be_embed= (const float*)d_in[5];
  const float* w_lin1  = (const float*)d_in[6];
  const float* g_lin1  = (const float*)d_in[7];
  const float* be_lin1 = (const float*)d_in[8];
  const float* down_w  = (const float*)d_in[9];
  const float* down_b  = (const float*)d_in[10];
  const float* gd      = (const float*)d_in[11];
  const float* bd      = (const float*)d_in[12];
  const float* up_w    = (const float*)d_in[13];
  const float* up_b    = (const float*)d_in[14];
  const float* gu      = (const float*)d_in[15];
  const float* bu      = (const float*)d_in[16];

  // workspace bump allocator (256B aligned)
  char* p = (char*)d_ws;
  auto alloc = [&](size_t bytes) -> void* {
    void* q = p; p += (bytes + 255) & ~(size_t)255; return q;
  };
  __bf16* f0   = (__bf16*)alloc((size_t)BATCH*NPTS*PREVD*2);
  int*    sel  = (int*)   alloc((size_t)BATCH*GRP*4);
  float*  lcx  = (float*) alloc((size_t)BATCH*GRP*3*4);
  int*    nn   = (int*)   alloc((size_t)BATCH*GRP*KNN*4);
  float*  relr = (float*) alloc((size_t)RROWS*3*4);
  float*  part = (float*) alloc((size_t)(RROWS/8)*2*4);
  float*  stats= (float*) alloc(256);
  float*  wgt  = (float*) alloc((size_t)RROWS*4);
  __bf16* E    = (__bf16*)alloc((size_t)RROWS*CPAD*2);
  __bf16* W1   = (__bf16*)alloc((size_t)OUTD*CPAD*2);
  __bf16* Wd   = (__bf16*)alloc((size_t)2*MIDD*OUTD*2);
  __bf16* Wu   = (__bf16*)alloc((size_t)2*OUTD*MIDD*2);
  float*  x    = (float*) alloc((size_t)RROWS*OUTD*4);
  float*  h    = (float*) alloc((size_t)RROWS*MIDD*4);

  k_embed<<<BATCH*NPTS/2, 256, 0, stream>>>(feat, w_embed, b_embed, g_embed, be_embed, f0);

  const size_t fps_lds = (size_t)(NPTS*3 + NPTS + 1024)*4 + 1024*4;
  k_fps<<<BATCH, 1024, fps_lds, stream>>>(xyz, sel);

  k_knn<<<BATCH*GRP, 256, 0, stream>>>(xyz, sel, lcx, nn);

  k_cvtw<<<(OUTD*CPAD + 255)/256, 256, 0, stream>>>(w_lin1, down_w, up_w, W1, Wd, Wu);

  k_expand<<<RROWS/8, 256, 0, stream>>>(xyz, f0, sel, lcx, nn, E, relr, part);
  k_stats<<<1, 256, 0, stream>>>(part, stats);
  k_weights<<<RROWS/256, 256, 0, stream>>>(relr, lcx, stats, wgt);

  // x = relu(bn(E @ W1^T)) * w
  k_gemm<CPAD, CPAD, OUTD, 0><<<dim3(RROWS/128, OUTD/64), 256, 0, stream>>>(
      E, nullptr, W1, g_lin1, be_lin1, wgt, x, nullptr);

  for (int i = 0; i < 2; ++i) {
    // h = relu(bn(x @ down^T + db))
    k_gemm<OUTD, OUTD, MIDD, 1><<<dim3(RROWS/128, MIDD/64), 256, 0, stream>>>(
        nullptr, x, Wd + (size_t)i*MIDD*OUTD, gd + i*MIDD, bd + i*MIDD,
        down_b + i*MIDD, h, nullptr);
    // x = relu(bn(h @ up^T + ub) + x)
    k_gemm<MIDD, MIDD, OUTD, 2><<<dim3(RROWS/128, OUTD/64), 256, 0, stream>>>(
        nullptr, h, Wu + (size_t)i*OUTD*MIDD, gu + i*OUTD, bu + i*OUTD,
        up_b + i*OUTD, x, x);
  }

  k_final<<<BATCH*GRP, 256, 0, stream>>>(x, (float*)d_out);
}
